// MultiheadSelfAttention_24309514895797
// MI455X (gfx1250) — matrix operand order repaired and checked by reading, not compiled, not run on hardware
//
#include <hip/hip_runtime.h>

#define BATCH  4
#define SEQ    2048
#define DMODEL 1024
#define HEADS  16
#define DHEAD  64
#define MROWS  (BATCH * SEQ)   // 8192

typedef __attribute__((ext_vector_type(16))) __bf16 bf16x16;
typedef __attribute__((ext_vector_type(8)))  float  f32x8;
typedef unsigned short u16;

union Frag {
  bf16x16 v;
  uint4   u[2];
};

static __device__ __forceinline__ u16 f2bf(float f) {
  union { float f; unsigned int u; } a;
  a.f = f;
  unsigned int u = a.u;
  u += 0x7FFFu + ((u >> 16) & 1u);   // round-to-nearest-even
  return (u16)(u >> 16);
}

// ---- CDNA5 async global->LDS DMA (no VGPR data movement, ASYNCcnt tracked)
static __device__ __forceinline__ void async_b128(void* lds_dst, const void* gsrc) {
  const unsigned lds_off = (unsigned)(unsigned long long)lds_dst; // low 32b of flat LDS addr = wave-relative LDS offset
  asm volatile("global_load_async_to_lds_b128 %0, %1, off"
               :: "v"(lds_off), "v"(gsrc) : "memory");
}
#define S_WAIT_ASYNC(imm) asm volatile("s_wait_asynccnt " imm ::: "memory")

// ---------------------------------------------------------------- cast kernel
__global__ void cvt_bf16_kernel(const float* __restrict__ src,
                                u16* __restrict__ dst, int n, float scale) {
  int i = blockIdx.x * blockDim.x + threadIdx.x;
  if (i < n) dst[i] = f2bf(src[i] * scale);
}

// ------------------------------------------------------- GEMM: C = A * W^T
// A: bf16 [M,1024] row-major.  W: bf16 [1024,1024] row-major (row = out feature).
// Block: 256 threads = 8 waves; block tile 256(M) x 64(N); wave tile 32x64.
// Weight tiles staged via async DMA into double-buffered LDS (shared by 8 waves).
// MODE 0: bf16 out row-major [M,1024]
// MODE 1: bf16 out transposed-per-head: Vt[B,H,DHEAD,SEQ]
// MODE 2: f32 out row-major [M,1024]
template <int MODE>
__global__ __launch_bounds__(256) void gemm_xw_kernel(
    const u16* __restrict__ A, const u16* __restrict__ W,
    u16* __restrict__ outb, float* __restrict__ outf) {
  __shared__ __align__(16) u16 wtile[2][64][32];   // [buf][n-row][k] : 4KB each

  const int tid  = threadIdx.x;
  const int lane = tid & 31;
  const int wave = tid >> 5;
  const int half = lane >> 4;
  const int lr   = lane & 15;
  const int m0   = blockIdx.y * 256 + wave * 32;   // wave's 32 M-rows
  const int n0   = blockIdx.x * 64;                // block's 64 N-cols

  // staging assignment: 64x32 tile = 256 x b128 chunks, 1 per thread
  const int srow = tid >> 2;          // 0..63
  const int scol = (tid & 3) * 8;     // 0,8,16,24

  f32x8 acc[2][4] = {};

  const u16* arow0 = A + (size_t)(m0 + lr) * DMODEL;
  const u16* arow1 = A + (size_t)(m0 + 16 + lr) * DMODEL;

  // prologue: stage K-step 0
  async_b128(&wtile[0][srow][scol],
             W + (size_t)(n0 + srow) * DMODEL + scol);

  for (int it = 0; it < DMODEL / 32; ++it) {
    const int kt  = it * 32;
    const int buf = it & 1;
    if (it + 1 < DMODEL / 32) {
      async_b128(&wtile[buf ^ 1][srow][scol],
                 W + (size_t)(n0 + srow) * DMODEL + kt + 32 + scol);
      S_WAIT_ASYNC("0x1");   // <=1 outstanding -> current tile landed
    } else {
      S_WAIT_ASYNC("0x0");
    }
    __syncthreads();

    Frag a0, a1;
    a0.u[0] = *(const uint4*)(arow0 + kt + half * 8);
    a0.u[1] = *(const uint4*)(arow0 + kt + 16 + half * 8);
    a1.u[0] = *(const uint4*)(arow1 + kt + half * 8);
    a1.u[1] = *(const uint4*)(arow1 + kt + 16 + half * 8);
#pragma unroll
    for (int j = 0; j < 4; ++j) {
      const u16* wr = &wtile[buf][j * 16 + lr][half * 8];
      Frag bf;
      bf.u[0] = *(const uint4*)(wr);
      bf.u[1] = *(const uint4*)(wr + 16);
      acc[0][j] = __builtin_amdgcn_wmma_f32_16x16x32_bf16(
          false, a0.v, false, bf.v, (short)0, acc[0][j], false, false);
      acc[1][j] = __builtin_amdgcn_wmma_f32_16x16x32_bf16(
          false, a1.v, false, bf.v, (short)0, acc[1][j], false, false);
    }
    __syncthreads();   // everyone done reading buf before it is re-staged
  }

#pragma unroll
  for (int r = 0; r < 2; ++r) {
#pragma unroll
    for (int j = 0; j < 4; ++j) {
#pragma unroll
      for (int v = 0; v < 8; ++v) {
        const int gm = m0 + r * 16 + half * 8 + v;
        const int gn = n0 + j * 16 + lr;
        const float val = acc[r][j][v];
        if constexpr (MODE == 0) {
          outb[(size_t)gm * DMODEL + gn] = f2bf(val);
        } else if constexpr (MODE == 1) {
          const int b = gm >> 11, s = gm & (SEQ - 1);
          const int h = gn >> 6, d = gn & (DHEAD - 1);
          outb[(((size_t)b * HEADS + h) * DHEAD + d) * SEQ + s] = f2bf(val);
        } else {
          outf[(size_t)gm * DMODEL + gn] = val;
        }
      }
    }
  }
}

// ------------------------------------------------- causal flash attention
// Qb: [B,SEQ,DMODEL] bf16 (pre-scaled by 1/sqrt(dh));  Kb: [B,SEQ,DMODEL] bf16
// Vt: [B,H,DHEAD,SEQ] bf16;  Ob: [B,SEQ,DMODEL] bf16
// Block: 128 threads = 4 waves, q-tile 64 rows; K/V^T tiles (32 keys) staged via
// async DMA into double-buffered LDS shared by the 4 waves.
__global__ __launch_bounds__(128) void attn_kernel(
    const u16* __restrict__ Qb, const u16* __restrict__ Kb,
    const u16* __restrict__ Vt, u16* __restrict__ Ob) {
  __shared__ __align__(16) u16 ktile[2][32][64];   // [buf][key][dh]  4KB each
  __shared__ __align__(16) u16 vtile[2][64][32];   // [buf][dh][key]  4KB each
  __shared__ __align__(16) u16 plds[4][16][32];    // per-wave P bounce tile

  const int tid  = threadIdx.x;
  const int lane = tid & 31;
  const int wave = tid >> 5;
  const int half = lane >> 4;
  const int lr   = lane & 15;
  const int bh = blockIdx.y;
  const int b = bh >> 4, h = bh & 15;
  const int qb = blockIdx.x * 64;
  const int q0 = qb + wave * 16;                   // this wave's 16 q rows

  // staging assignment (per thread, 2 b128 chunks for K, 2 for V)
  const int kr0 = tid >> 3, kc0 = (tid & 7) * 8;            // chunk tid
  const int kr1 = (tid + 128) >> 3, kc1 = ((tid + 128) & 7) * 8;
  const int vr0 = tid >> 2, vc0 = (tid & 3) * 8;
  const int vr1 = (tid + 128) >> 2, vc1 = ((tid + 128) & 3) * 8;

  const u16* Kbase = Kb + (size_t)b * SEQ * DMODEL + h * DHEAD;
  const u16* Vbase = Vt + ((size_t)b * HEADS + h) * DHEAD * SEQ;

  // Q A-fragments for dh 0..31 and 32..63 (registers, whole kernel)
  const u16* qrow = Qb + ((size_t)b * SEQ + q0 + lr) * DMODEL + h * DHEAD;
  Frag aq0, aq1;
  aq0.u[0] = *(const uint4*)(qrow + half * 8);
  aq0.u[1] = *(const uint4*)(qrow + 16 + half * 8);
  aq1.u[0] = *(const uint4*)(qrow + 32 + half * 8);
  aq1.u[1] = *(const uint4*)(qrow + 48 + half * 8);

  f32x8 o[4] = {};
  float mst[8], lst[8];
#pragma unroll
  for (int v = 0; v < 8; ++v) { mst[v] = -1e30f; lst[v] = 0.0f; }

  const int ntiles = (qb + 64) >> 5;   // 32-key tiles needed by this block

  // prologue: stage tile 0
  {
    async_b128(&ktile[0][kr0][kc0], Kbase + (size_t)kr0 * DMODEL + kc0);
    async_b128(&ktile[0][kr1][kc1], Kbase + (size_t)kr1 * DMODEL + kc1);
    async_b128(&vtile[0][vr0][vc0], Vbase + (size_t)vr0 * SEQ + vc0);
    async_b128(&vtile[0][vr1][vc1], Vbase + (size_t)vr1 * SEQ + vc1);
  }

  for (int it = 0; it < ntiles; ++it) {
    const int k0  = it * 32;
    const int buf = it & 1;
    if (it + 1 < ntiles) {
      const int kn = k0 + 32;
      async_b128(&ktile[buf ^ 1][kr0][kc0], Kbase + (size_t)(kn + kr0) * DMODEL + kc0);
      async_b128(&ktile[buf ^ 1][kr1][kc1], Kbase + (size_t)(kn + kr1) * DMODEL + kc1);
      async_b128(&vtile[buf ^ 1][vr0][vc0], Vbase + (size_t)vr0 * SEQ + kn + vc0);
      async_b128(&vtile[buf ^ 1][vr1][vc1], Vbase + (size_t)vr1 * SEQ + kn + vc1);
      S_WAIT_ASYNC("0x4");   // <=4 outstanding -> current tile landed
    } else {
      S_WAIT_ASYNC("0x0");
    }
    __syncthreads();

    if (k0 < q0 + 16) {      // wave-uniform causal gate; EXEC stays all-ones
      // ---- scores S = Q * K^T  (16 q-rows x 32 keys)
      f32x8 s[2];
#pragma unroll
      for (int t = 0; t < 2; ++t) {
        const u16* kr = &ktile[buf][t * 16 + lr][half * 8];
        Frag bk0, bk1;
        bk0.u[0] = *(const uint4*)(kr);
        bk0.u[1] = *(const uint4*)(kr + 16);
        bk1.u[0] = *(const uint4*)(kr + 32);
        bk1.u[1] = *(const uint4*)(kr + 48);
        f32x8 z = {};
        z = __builtin_amdgcn_wmma_f32_16x16x32_bf16(
            false, aq0.v, false, bk0.v, (short)0, z, false, false);
        s[t] = __builtin_amdgcn_wmma_f32_16x16x32_bf16(
            false, aq1.v, false, bk1.v, (short)0, z, false, false);
      }

      // ---- causal mask
      if (k0 + 31 > q0) {
#pragma unroll
        for (int t = 0; t < 2; ++t) {
          const int key = k0 + t * 16 + lr;
#pragma unroll
          for (int v = 0; v < 8; ++v) {
            const int row = q0 + half * 8 + v;
            if (key > row) s[t][v] = -1e30f;
          }
        }
      }

      // ---- online softmax row stats (rows live across 16-lane halves)
      float alpha[8];
#pragma unroll
      for (int v = 0; v < 8; ++v) {
        float tm = fmaxf(s[0][v], s[1][v]);
        tm = fmaxf(tm, __shfl_xor(tm, 1, 32));
        tm = fmaxf(tm, __shfl_xor(tm, 2, 32));
        tm = fmaxf(tm, __shfl_xor(tm, 4, 32));
        tm = fmaxf(tm, __shfl_xor(tm, 8, 32));
        const float mnew = fmaxf(mst[v], tm);
        alpha[v] = __expf(mst[v] - mnew);
        mst[v] = mnew;
      }
#pragma unroll
      for (int v = 0; v < 8; ++v) {
        const float p0 = __expf(s[0][v] - mst[v]);
        const float p1 = __expf(s[1][v] - mst[v]);
        s[0][v] = p0; s[1][v] = p1;
        float rs = p0 + p1;
        rs += __shfl_xor(rs, 1, 32);
        rs += __shfl_xor(rs, 2, 32);
        rs += __shfl_xor(rs, 4, 32);
        rs += __shfl_xor(rs, 8, 32);
        lst[v] = lst[v] * alpha[v] + rs;
#pragma unroll
        for (int j = 0; j < 4; ++j) o[j][v] *= alpha[v];
      }

      // ---- re-layout P (C/D frag) -> A frag via per-wave LDS tile
      //      (intra-wave DS ops are in-order on CDNA5)
#pragma unroll
      for (int t = 0; t < 2; ++t)
#pragma unroll
        for (int v = 0; v < 8; ++v)
          plds[wave][half * 8 + v][t * 16 + lr] = f2bf(s[t][v]);

      Frag pa;
      const u16* prow = &plds[wave][lr][0];
      pa.u[0] = *(const uint4*)(prow + half * 8);
      pa.u[1] = *(const uint4*)(prow + 16 + half * 8);

      // ---- O += P * V
#pragma unroll
      for (int j = 0; j < 4; ++j) {
        const u16* vr = &vtile[buf][j * 16 + lr][half * 8];
        Frag bv;
        bv.u[0] = *(const uint4*)(vr);
        bv.u[1] = *(const uint4*)(vr + 16);
        o[j] = __builtin_amdgcn_wmma_f32_16x16x32_bf16(
            false, pa.v, false, bv.v, (short)0, o[j], false, false);
      }
    }
    __syncthreads();   // all waves done with buf before it is re-staged
  }

  // ---- normalize and store O as bf16 [B,SEQ,DMODEL]
#pragma unroll
  for (int v = 0; v < 8; ++v) {
    const float rinv = 1.0f / lst[v];
    const int m = q0 + half * 8 + v;
#pragma unroll
    for (int j = 0; j < 4; ++j) {
      const int n = h * DHEAD + j * 16 + lr;
      Ob[((size_t)b * SEQ + m) * DMODEL + n] = f2bf(o[j][v] * rinv);
    }
  }
}

// ---------------------------------------------------------------- launcher
extern "C" void kernel_launch(void* const* d_in, const int* in_sizes, int n_in,
                              void* d_out, int out_size, void* d_ws, size_t ws_size,
                              hipStream_t stream) {
  const float* x  = (const float*)d_in[0];
  const float* wq = (const float*)d_in[1];
  const float* wk = (const float*)d_in[2];
  const float* wv = (const float*)d_in[3];
  const float* wo = (const float*)d_in[4];
  float* out = (float*)d_out;

  const size_t NX = (size_t)MROWS * DMODEL;     // 8M elems
  const size_t NW = (size_t)DMODEL * DMODEL;    // 1M elems

  char* ws = (char*)d_ws;
  u16* xb  = (u16*)ws;                   ws += NX * 2;
  u16* wqb = (u16*)ws;                   ws += NW * 2;
  u16* wkb = (u16*)ws;                   ws += NW * 2;
  u16* wvb = (u16*)ws;                   ws += NW * 2;
  u16* wob = (u16*)ws;                   ws += NW * 2;
  u16* Qb  = (u16*)ws;                   ws += NX * 2;
  u16* Kb  = (u16*)ws;                   ws += NX * 2;
  u16* Vt  = (u16*)ws;                   ws += NX * 2;
  u16* Ob  = (u16*)ws;                   ws += NX * 2;   // total ~88 MB

  const float qscale = 0.125f;  // 1/sqrt(DHEAD), folded into Wq

  // 1) cast to bf16
  cvt_bf16_kernel<<<(int)((NX + 255) / 256), 256, 0, stream>>>(x, xb, (int)NX, 1.0f);
  cvt_bf16_kernel<<<(int)((NW + 255) / 256), 256, 0, stream>>>(wq, wqb, (int)NW, qscale);
  cvt_bf16_kernel<<<(int)((NW + 255) / 256), 256, 0, stream>>>(wk, wkb, (int)NW, 1.0f);
  cvt_bf16_kernel<<<(int)((NW + 255) / 256), 256, 0, stream>>>(wv, wvb, (int)NW, 1.0f);
  cvt_bf16_kernel<<<(int)((NW + 255) / 256), 256, 0, stream>>>(wo, wob, (int)NW, 1.0f);

  // 2) projections (M=8192 x N=1024 x K=1024, bf16 WMMA, async-staged weights)
  const dim3 ggrid(DMODEL / 64, MROWS / 256);
  gemm_xw_kernel<0><<<ggrid, 256, 0, stream>>>(xb, wqb, Qb, nullptr);
  gemm_xw_kernel<0><<<ggrid, 256, 0, stream>>>(xb, wkb, Kb, nullptr);
  gemm_xw_kernel<1><<<ggrid, 256, 0, stream>>>(xb, wvb, Vt, nullptr);

  // 3) causal flash attention (async-staged K/V^T)
  attn_kernel<<<dim3(SEQ / 64, BATCH * HEADS), 128, 0, stream>>>(Qb, Kb, Vt, Ob);

  // 4) output projection -> f32 d_out
  gemm_xw_kernel<2><<<ggrid, 256, 0, stream>>>(Ob, wob, nullptr, out);

  (void)in_sizes; (void)n_in; (void)out_size; (void)ws_size;
}
